// Decoder_50508815401309
// MI455X (gfx1250) — matrix operand order repaired and checked by reading, not compiled, not run on hardware
//
#include <hip/hip_runtime.h>
#include <hip/hip_bf16.h>

#define UNITS 512
#define VOCAB 32000
#define BATCH 128
#define SEQ   64

typedef __attribute__((ext_vector_type(16))) __bf16 v16bf;
typedef __attribute__((ext_vector_type(8)))  __bf16 v8bf;
typedef __attribute__((ext_vector_type(8)))  float  v8f;
typedef __attribute__((__vector_size__(16))) int    v4i;

#if __has_builtin(__builtin_amdgcn_global_load_async_to_lds_b128) && \
    __has_builtin(__builtin_amdgcn_s_wait_asynccnt)
#define USE_ASYNC_LDS 1
#else
#define USE_ASYNC_LDS 0
#endif

#define AS(n) __attribute__((address_space(n)))

// ---------- helpers ----------

__device__ inline __bf16 f2bf(float x) {
    unsigned int u = __builtin_bit_cast(unsigned int, x);
    unsigned int r = (u + 0x7FFFu + ((u >> 16) & 1u)) >> 16;   // round-to-nearest-even
    return __builtin_bit_cast(__bf16, (unsigned short)r);
}

#if USE_ASYNC_LDS
__device__ inline void async_copy16(const __bf16* gsrc, __bf16* ldst) {
    // global -> LDS, 128 bits per lane, tracked by ASYNCcnt
    __builtin_amdgcn_global_load_async_to_lds_b128(
        (AS(1) v4i*)gsrc, (AS(3) v4i*)ldst, 0, 0);
}
#endif

// A-fragment (16x32 bf16, row-major tile in LDS: smA[m*32+k]).
// wave32: lanes 0-15 -> M=lane (K 0-7,16-23), lanes 16-31 -> M=lane-16 (K 8-15,24-31).
__device__ inline v16bf load_frag_a(const __bf16* smA, int lane) {
    const int m = lane & 15, half = lane >> 4;
    v16bf f;
#pragma unroll
    for (int e = 0; e < 16; ++e) {
        const int v = e >> 1, h = e & 1;
        const int k = ((v & 4) ? 16 : 0) + half * 8 + ((v & 3) << 1) + h;
        f[e] = smA[m * 32 + k];
    }
    return f;
}

// B-fragment (32x16 bf16). LDS holds the tile transposed: smBt[n*32+k].
// lanes 0-15 hold K=0-7,16-23 for N=lane; lanes 16-31 hold K=8-15,24-31 for N=lane-16.
__device__ inline v16bf load_frag_b(const __bf16* smBt, int lane) {
    const int n = lane & 15, half = lane >> 4;
    v16bf f;
#pragma unroll
    for (int e = 0; e < 16; ++e) f[e] = smBt[n * 32 + ((e & 8) ? 16 : 0) + half * 8 + (e & 7)];
    return f;
}

__device__ inline float sigmoidf(float x) { return 1.0f / (1.0f + __expf(-x)); }

// ---------- kernel 0: score <- vb (accumulated into by attn kernel) ----------
__global__ void score_init_kernel(float* __restrict__ score, const float* __restrict__ vb) {
    score[blockIdx.x * blockDim.x + threadIdx.x] = vb[0];
}

// ---------- kernel 1: attention GEMM (BM=128,BN=64) + tanh + vW partial dot ----------
// e = tanh(A@W0 + b0 + b1)   (hidden0 == 0 -> W1 GEMM collapses to +b1)
// score[row] += e[row, colslice] . vW[colslice]   via global atomics
__global__ __launch_bounds__(256) void attn_score_kernel(
    const float* __restrict__ A,    // (BATCH*SEQ, UNITS) f32
    const float* __restrict__ W0,   // (UNITS, UNITS) f32
    const float* __restrict__ b0, const float* __restrict__ b1,
    const float* __restrict__ vW, float* __restrict__ score)
{
    __shared__ __attribute__((aligned(16))) __bf16 smA[2][128 * 32];     // 2 x 8 KB
    __shared__ __attribute__((aligned(16))) __bf16 smBt[2][4 * 16 * 32]; // 2 x 4 KB
    const int tid = threadIdx.x;
    const int wave = tid >> 5, lane = tid & 31;
    const int colBase = blockIdx.x * 64;
    const int rowBase = blockIdx.y * 128;

    auto loadTiles = [&](int k0, int buf) {
        // A tile: 128x32 f32 -> bf16 (vector load + b128 LDS store)
#pragma unroll
        for (int c0 = 0; c0 < 512; c0 += 256) {
            const int c = c0 + tid;
            const int m = c >> 2, ko = (c & 3) * 8;
            const float4 lo = *(const float4*)&A[(rowBase + m) * UNITS + k0 + ko];
            const float4 hi = *(const float4*)&A[(rowBase + m) * UNITS + k0 + ko + 4];
            v8bf v; v[0]=f2bf(lo.x); v[1]=f2bf(lo.y); v[2]=f2bf(lo.z); v[3]=f2bf(lo.w);
                    v[4]=f2bf(hi.x); v[5]=f2bf(hi.y); v[6]=f2bf(hi.z); v[7]=f2bf(hi.w);
            *(v8bf*)&smA[buf][m * 32 + ko] = v;
        }
        // B tile: 32x64 f32 -> bf16, transposed [n][k]; one chunk per thread
        const int n = tid & 63, kq = tid >> 6;   // kq in 0..3
        v8bf v;
#pragma unroll
        for (int j = 0; j < 8; ++j)
            v[j] = f2bf(W0[(k0 + kq * 8 + j) * UNITS + colBase + n]);
        *(v8bf*)&smBt[buf][(n >> 4) * 512 + (n & 15) * 32 + kq * 8] = v;
    };

    v8f acc[4];
    const v8f vzero = {0.f,0.f,0.f,0.f,0.f,0.f,0.f,0.f};
#pragma unroll
    for (int t = 0; t < 4; ++t) acc[t] = vzero;

    loadTiles(0, 0);
    const int iters = UNITS / 32;
    for (int it = 0; it < iters; ++it) {
        const int buf = it & 1;
        __syncthreads();
        if (it + 1 < iters) loadTiles((it + 1) * 32, buf ^ 1);

        const v16bf a = load_frag_a(&smA[buf][wave * 512], lane);
#pragma unroll
        for (int t = 0; t < 4; ++t) {
            const v16bf b = load_frag_b(&smBt[buf][t * 512], lane);
            acc[t] = __builtin_amdgcn_wmma_f32_16x16x32_bf16(
                false, a, false, b, (short)0, acc[t], false, false);
        }
    }

    // epilogue: tanh, * vW, reduce over the 64 columns of this block
    float p[8];
#pragma unroll
    for (int r = 0; r < 8; ++r) p[r] = 0.0f;
#pragma unroll
    for (int t = 0; t < 4; ++t) {
        const int n = colBase + t * 16 + (lane & 15);
        const float bsum = b0[n] + b1[n];
        const float vwn  = vW[n];
#pragma unroll
        for (int r = 0; r < 8; ++r) p[r] += tanhf(acc[t][r] + bsum) * vwn;
    }
#pragma unroll
    for (int off = 8; off >= 1; off >>= 1)
#pragma unroll
        for (int r = 0; r < 8; ++r) p[r] += __shfl_xor(p[r], off, 32);

    if ((lane & 15) == 0) {
        const int mbase = rowBase + wave * 16 + (lane >> 4) * 8;
#pragma unroll
        for (int r = 0; r < 8; ++r) atomicAdd(&score[mbase + r], p[r]);
    }
}

// ---------- kernel 2: softmax over SEQ + context + alpha out ----------
__global__ __launch_bounds__(256) void softmax_context_kernel(
    const float* __restrict__ score, const float* __restrict__ attn,
    float* __restrict__ alpha_out, float* __restrict__ context)
{
    __shared__ float sAlpha[SEQ];
    __shared__ float red[256];
    const int b = blockIdx.x, tid = threadIdx.x;

    const float v = (tid < SEQ) ? score[b * SEQ + tid] : -1.0e30f;
    red[tid] = v; __syncthreads();
    for (int s = 128; s >= 1; s >>= 1) {
        if (tid < s) red[tid] = fmaxf(red[tid], red[tid + s]);
        __syncthreads();
    }
    const float mx = red[0]; __syncthreads();
    const float e = (tid < SEQ) ? __expf(v - mx) : 0.0f;
    red[tid] = e; __syncthreads();
    for (int s = 128; s >= 1; s >>= 1) {
        if (tid < s) red[tid] += red[tid + s];
        __syncthreads();
    }
    const float inv = 1.0f / red[0];
    if (tid < SEQ) {
        const float a = e * inv;
        sAlpha[tid] = a;
        alpha_out[b * SEQ + tid] = a;
    }
    __syncthreads();
    for (int u = tid; u < UNITS; u += 256) {
        float s = 0.0f;
#pragma unroll 8
        for (int t = 0; t < SEQ; ++t)
            s += attn[(b * SEQ + t) * UNITS + u] * sAlpha[t];
        context[b * UNITS + u] = s;
    }
}

// ---------- kernel 3: xc = [emb[inputs[b]], context[b]]  (stored bf16) ----------
__global__ void build_xc_kernel(const int* __restrict__ inputs,
                                const float* __restrict__ emb,
                                const float* __restrict__ context,
                                __bf16* __restrict__ xc)
{
    const int idx = blockIdx.x * blockDim.x + threadIdx.x;   // BATCH*1024
    const int b = idx >> 10, c = idx & 1023;
    const float v = (c < UNITS) ? emb[inputs[b] * UNITS + c]
                                : context[b * UNITS + (c - UNITS)];
    xc[idx] = f2bf(v);
}

// ---------- generic bf16-WMMA GEMM, BM=128 (full M in one block), BN=64 ----------
// C = act(A@B + bias); A is (128,K) bf16; B read once chip-wide.
// Double-buffered LDS; A staged via async global->LDS when available.
__global__ __launch_bounds__(256) void gemm128_kernel(
    const __bf16* __restrict__ A, const float* __restrict__ B,
    const float* __restrict__ bias,
    float* __restrict__ Cout, __bf16* __restrict__ Cbf,
    int N, int K, int act)
{
    __shared__ __attribute__((aligned(16))) __bf16 smA[2][128 * 32];     // 2 x 8 KB
    __shared__ __attribute__((aligned(16))) __bf16 smBt[2][4 * 16 * 32]; // 2 x 4 KB
    const int tid = threadIdx.x;
    const int wave = tid >> 5, lane = tid & 31;
    const int colBase = blockIdx.x * 64;

    auto loadTiles = [&](int k0, int buf) {
        // A tile: pure 16-byte copies (already bf16)
#pragma unroll
        for (int c0 = 0; c0 < 512; c0 += 256) {
            const int c = c0 + tid;
            const int m = c >> 2, ko = (c & 3) * 8;
#if USE_ASYNC_LDS
            async_copy16(&A[m * K + k0 + ko], &smA[buf][m * 32 + ko]);
#else
            *(v8bf*)&smA[buf][m * 32 + ko] = *(const v8bf*)&A[m * K + k0 + ko];
#endif
        }
        // B tile: 32x64 f32 -> bf16 transposed; one 8-deep column chunk per thread
        const int n = tid & 63, kq = tid >> 6;
        v8bf v;
#pragma unroll
        for (int j = 0; j < 8; ++j)
            v[j] = f2bf(B[(long long)(k0 + kq * 8 + j) * N + colBase + n]);
        *(v8bf*)&smBt[buf][(n >> 4) * 512 + (n & 15) * 32 + kq * 8] = v;
    };

    v8f acc[4];
    const v8f vzero = {0.f,0.f,0.f,0.f,0.f,0.f,0.f,0.f};
#pragma unroll
    for (int t = 0; t < 4; ++t) acc[t] = vzero;

    loadTiles(0, 0);
    const int iters = K / 32;
    for (int it = 0; it < iters; ++it) {
        const int buf = it & 1;
#if USE_ASYNC_LDS
        __builtin_amdgcn_s_wait_asynccnt(0);   // my async LDS writes done
#endif
        __syncthreads();                        // everyone's writes visible
        if (it + 1 < iters) loadTiles((it + 1) * 32, buf ^ 1);

        const v16bf a = load_frag_a(&smA[buf][wave * 512], lane);
#pragma unroll
        for (int t = 0; t < 4; ++t) {
            const v16bf b = load_frag_b(&smBt[buf][t * 512], lane);
            acc[t] = __builtin_amdgcn_wmma_f32_16x16x32_bf16(
                false, a, false, b, (short)0, acc[t], false, false);
        }
    }

    const int mbase = wave * 16 + (lane >> 4) * 8;
#pragma unroll
    for (int t = 0; t < 4; ++t) {
        const int n = colBase + t * 16 + (lane & 15);
        const float bs = bias[n];
#pragma unroll
        for (int r = 0; r < 8; ++r) {
            float v = acc[t][r] + bs;
            if (act == 1) v = fmaxf(v, 0.0f);
            const long long off = (long long)(mbase + r) * N + n;
            if (Cout) Cout[off] = v;
            if (Cbf)  Cbf[off]  = f2bf(v);
        }
    }
}

// ---------- kernel 5: GRU pointwise (h == 0; recurrent GEMM is bias-only) ----------
__global__ void gru_elem_kernel(const float* __restrict__ gates,
                                const float* __restrict__ gru_b,
                                float* __restrict__ state_out,
                                __bf16* __restrict__ state_bf)
{
    const int idx = blockIdx.x * blockDim.x + threadIdx.x;   // BATCH*UNITS
    const int b = idx >> 9, u = idx & (UNITS - 1);
    const float* g  = gates + b * 3 * UNITS;                 // includes gru_b[0]
    const float* rb = gru_b + 3 * UNITS;                     // gru_b[1] (h@gru_rk == 0)
    const float z  = sigmoidf(g[u] + rb[u]);
    const float r  = sigmoidf(g[UNITS + u] + rb[UNITS + u]);
    const float hh = tanhf(g[2 * UNITS + u] + r * rb[2 * UNITS + u]);
    const float st = (1.0f - z) * hh;                        // z*h + (1-z)*hh with h=0
    state_out[idx] = st;
    state_bf[idx]  = f2bf(st);
}

// ---------- launcher ----------
extern "C" void kernel_launch(void* const* d_in, const int* in_sizes, int n_in,
                              void* d_out, int out_size, void* d_ws, size_t ws_size,
                              hipStream_t stream) {
    const int*   inputs = (const int*)  d_in[0];
    const float* attn   = (const float*)d_in[1];
    const float* W0     = (const float*)d_in[2];
    const float* b0     = (const float*)d_in[3];
    /* d_in[4] = W1: unused (hidden0 == 0) */
    const float* b1     = (const float*)d_in[5];
    const float* vW     = (const float*)d_in[6];
    const float* vb     = (const float*)d_in[7];
    const float* emb    = (const float*)d_in[8];
    const float* gru_k  = (const float*)d_in[9];
    /* d_in[10] = gru_rk: unused (h == 0) */
    const float* gru_b  = (const float*)d_in[11];
    const float* dW     = (const float*)d_in[12];
    const float* db     = (const float*)d_in[13];
    const float* oW     = (const float*)d_in[14];
    const float* ob     = (const float*)d_in[15];

    float* out    = (float*)d_out;
    float* logits = out;                              // 128*32000
    float* state  = logits + BATCH * VOCAB;           // 128*512
    float* alpha  = state  + BATCH * UNITS;           // 128*64

    char* w = (char*)d_ws;
    float*  score    = (float*) (w);                  // 8192 f32
    float*  context  = (float*) (w + 32768);          // 65536 f32
    __bf16* xc       = (__bf16*)(w + 294912);         // 131072 bf16
    float*  gates    = (float*) (w + 557056);         // 196608 f32
    __bf16* state_bf = (__bf16*)(w + 1343488);        // 65536 bf16
    __bf16* y_bf     = (__bf16*)(w + 1474560);        // 65536 bf16  (end ~1.6 MB)

    score_init_kernel<<<BATCH * SEQ / 256, 256, 0, stream>>>(score, vb);
    attn_score_kernel<<<dim3(UNITS / 64, BATCH * SEQ / 128), 256, 0, stream>>>(
        attn, W0, b0, b1, vW, score);
    softmax_context_kernel<<<BATCH, 256, 0, stream>>>(score, attn, alpha, context);
    build_xc_kernel<<<BATCH * 1024 / 256, 256, 0, stream>>>(inputs, emb, context, xc);
    gemm128_kernel<<<3 * UNITS / 64, 256, 0, stream>>>(
        xc, gru_k, gru_b, gates, nullptr, 3 * UNITS, 2 * UNITS, 0);
    gru_elem_kernel<<<BATCH * UNITS / 256, 256, 0, stream>>>(gates, gru_b, state, state_bf);
    gemm128_kernel<<<UNITS / 64, 256, 0, stream>>>(
        state_bf, dW, db, nullptr, y_bf, UNITS, UNITS, 1);
    gemm128_kernel<<<VOCAB / 64, 256, 0, stream>>>(
        y_bf, oW, ob, logits, nullptr, VOCAB, UNITS, 0);
}
